// VectorMapNet_55602646614710
// MI455X (gfx1250) — matrix operand order repaired and checked by reading, not compiled, not run on hardware
//
#include <hip/hip_runtime.h>

typedef __attribute__((ext_vector_type(16))) _Float16 v16h;
typedef __attribute__((ext_vector_type(8)))  float    v8f;

#define TPB 256
static inline int G(long n){ return (int)((n + TPB - 1) / TPB); }

union Frag16 { float4 f4[2]; v16h h; };

// ======================= WMMA GEMM =======================
// C[n][m] = cscale * sum_k A[m][k]*B[n][k] + bias[m]
// A: [M][Kd] fp16 (row-major, K contiguous), B: [N][Kd] fp16, C: [N][M] fp32.
// Requires M%16==0, N%64==0, Kd%32==0.
// Each wave32 computes a 16x64 strip (1 A-frag reused over 4 B-frags / 4 accumulators).
// All 5 fragment loads are issued before the 4 WMMAs so the scheduler can use
// partial s_wait_loadcnt and overlap loads with matrix issue.
__global__ __launch_bounds__(256) void k_wmma_gemm(
    const _Float16* __restrict__ A, const _Float16* __restrict__ B,
    float* __restrict__ C, const float* __restrict__ bias,
    int M, int Kd, int N, float cscale,
    long long sA, long long sB, long long sC)
{
  int bt = blockIdx.y;
  A += (long long)bt * sA; B += (long long)bt * sB; C += (long long)bt * sC;
  int wave = (int)(threadIdx.x >> 5);
  int lane = (int)(threadIdx.x & 31);
  int mt = M >> 4, ng = N >> 6;              // ng = groups of 4 n-tiles
  long unit = (long)mt * ng;
  long t0 = (long)blockIdx.x * 8 + wave;
  if (t0 >= unit) return;                    // uniform per-wave exit, EXEC stays all-ones
  int tm = (int)(t0 % mt), tg = (int)(t0 / mt);
  int m0 = tm << 4, n0 = tg << 6;
  int l15 = lane & 15;
  int hi  = lane >> 4;
  // A fragment in this code's contiguous k order (same as B): lanes 0-15 hold K 0..15, lanes 16-31 hold K 16..31
  const _Float16* arow = A + (long)(m0 + l15) * Kd + hi * 16;
  // B fragment in the same contiguous k order as A: lanes 0-15 hold K 0..15, lanes 16-31 hold K 16..31 (col = lane&15)
  const _Float16* b0 = B + (long)(n0      + l15) * Kd + hi * 16;
  const _Float16* b1 = B + (long)(n0 + 16 + l15) * Kd + hi * 16;
  const _Float16* b2 = B + (long)(n0 + 32 + l15) * Kd + hi * 16;
  const _Float16* b3 = B + (long)(n0 + 48 + l15) * Kd + hi * 16;
  v8f zero = {0.f,0.f,0.f,0.f,0.f,0.f,0.f,0.f};
  v8f acc0 = zero, acc1 = zero, acc2 = zero, acc3 = zero;
  for (int k0 = 0; k0 < Kd; k0 += 32) {
    Frag16 fa, fb0, fb1, fb2, fb3;
    // batch all loads first (10 b128 loads in flight per k-step)
    fa.f4[0]  = *(const float4*)(arow + k0);
    fa.f4[1]  = *(const float4*)(arow + k0 + 8);
    fb0.f4[0] = *(const float4*)(b0 + k0);
    fb0.f4[1] = *(const float4*)(b0 + k0 + 8);
    fb1.f4[0] = *(const float4*)(b1 + k0);
    fb1.f4[1] = *(const float4*)(b1 + k0 + 8);
    fb2.f4[0] = *(const float4*)(b2 + k0);
    fb2.f4[1] = *(const float4*)(b2 + k0 + 8);
    fb3.f4[0] = *(const float4*)(b3 + k0);
    fb3.f4[1] = *(const float4*)(b3 + k0 + 8);
    acc0 = __builtin_amdgcn_wmma_f32_16x16x32_f16(false, fa.h, false, fb0.h,
                                                  (short)0, acc0, false, false);
    acc1 = __builtin_amdgcn_wmma_f32_16x16x32_f16(false, fa.h, false, fb1.h,
                                                  (short)0, acc1, false, false);
    acc2 = __builtin_amdgcn_wmma_f32_16x16x32_f16(false, fa.h, false, fb2.h,
                                                  (short)0, acc2, false, false);
    acc3 = __builtin_amdgcn_wmma_f32_16x16x32_f16(false, fa.h, false, fb3.h,
                                                  (short)0, acc3, false, false);
  }
  int mbase = m0 + hi * 8;
  v8f accs[4] = {acc0, acc1, acc2, acc3};
#pragma unroll
  for (int nb = 0; nb < 4; nb++){
    float o[8];
#pragma unroll
    for (int r = 0; r < 8; r++) o[r] = cscale * accs[nb][r] + (bias ? bias[mbase + r] : 0.f);
    float* crow = C + (long)(n0 + nb * 16 + l15) * M + mbase;
    *(float4*)(crow)     = make_float4(o[0], o[1], o[2], o[3]);
    *(float4*)(crow + 4) = make_float4(o[4], o[5], o[6], o[7]);
  }
}

// ======================= frontend kernels =======================
__global__ void k_cvt_w16(const float* __restrict__ Wsrc, _Float16* __restrict__ Wdst,
                          int M, int K, int Kp){
  int i = blockIdx.x * TPB + threadIdx.x;
  if (i >= M * Kp) return;
  int m = i / Kp, k = i % Kp;
  Wdst[i] = (k < K) ? (_Float16)Wsrc[(long)m * K + k] : (_Float16)0.f;
}

__global__ void k_cell_softmax(const float* __restrict__ vert, float* __restrict__ s,
                               int hc, int wc){
  int cells = hc * wc;
  int i = blockIdx.x * TPB + threadIdx.x;
  if (i >= 2 * cells) return;
  int b = i / cells, r = i % cells;
  int cy = r / wc, cx = r % wc;
  const float* p = vert + (long)b * 65 * cells + r;
  float mx = -3.4e38f;
  for (int c = 0; c < 65; c++) mx = fmaxf(mx, p[(long)c * cells]);
  float sum = 0.f;
  for (int c = 0; c < 65; c++) sum += expf(p[(long)c * cells] - mx);
  float inv = 1.f / sum;
  int Wf = wc * 8;
  long base = (long)b * (hc * 8) * (long)Wf;
  for (int c = 0; c < 64; c++){
    int y = cy * 8 + (c >> 3), x = cx * 8 + (c & 7);
    s[base + (long)y * Wf + x] = expf(p[(long)c * cells] - mx) * inv;
  }
}

__global__ void k_mp_row(const float* __restrict__ in, float* __restrict__ out, int Hf, int Wf){
  long i = (long)blockIdx.x * TPB + threadIdx.x;
  long npix = (long)Hf * Wf;
  if (i >= 2 * npix) return;
  int x = (int)(i % Wf);
  float m = -3.4e38f;
  for (int d = -4; d <= 4; d++){ int xx = x + d; if (xx >= 0 && xx < Wf) m = fmaxf(m, in[i + d]); }
  out[i] = m;
}
__global__ void k_mp_col(const float* __restrict__ in, float* __restrict__ out, int Hf, int Wf){
  long i = (long)blockIdx.x * TPB + threadIdx.x;
  long npix = (long)Hf * Wf;
  if (i >= 2 * npix) return;
  int y = (int)((i % npix) / Wf);
  float m = -3.4e38f;
  for (int d = -4; d <= 4; d++){ int yy = y + d; if (yy >= 0 && yy < Hf) m = fmaxf(m, in[i + (long)d * Wf]); }
  out[i] = m;
}
__global__ void k_eq_mask(const float* s, const float* pool, float* mask, long n){
  long i = (long)blockIdx.x * TPB + threadIdx.x; if (i >= n) return;
  mask[i] = (s[i] == pool[i]) ? 1.f : 0.f;
}
__global__ void k_suppress(const float* pm, const float* s, float* ss, long n){
  long i = (long)blockIdx.x * TPB + threadIdx.x; if (i >= n) return;
  ss[i] = (pm[i] > 0.f) ? 0.f : s[i];
}
__global__ void k_mask_update(float* mask, const float* ss, const float* pss, const float* pm, long n){
  long i = (long)blockIdx.x * TPB + threadIdx.x; if (i >= n) return;
  bool keep = mask[i] > 0.f;
  bool nm   = (ss[i] == pss[i]) && !(pm[i] > 0.f);
  mask[i] = (keep || nm) ? 1.f : 0.f;
}
__global__ void k_apply_mask(float* s, const float* mask, long n){
  long i = (long)blockIdx.x * TPB + threadIdx.x; if (i >= n) return;
  s[i] = (mask[i] > 0.f) ? s[i] : 0.f;
}

// ---- top-K: two-level radix select on float bits (values >= 0) + bitonic sort ----
__global__ void k_hist1(const float* s, int* hist, long npix){
  long i = (long)blockIdx.x * TPB + threadIdx.x;
  if (i >= 2 * npix) return;
  int b = (int)(i / npix);
  unsigned u = __float_as_uint(s[i]);
  atomicAdd(&hist[b * 256 + (int)(u >> 24)], 1);
}
__global__ void k_thresh1(const int* hist, int* meta, int Ksel){
  int b = blockIdx.x;
  const int* h = hist + b * 256;
  int cum = 0, T = 0;
  for (int bin = 255; bin >= 0; bin--){
    int c = h[bin];
    if (cum + c >= Ksel){ T = bin; break; }
    cum += c;
  }
  meta[b * 8 + 0] = T; meta[b * 8 + 1] = cum;
}
__global__ void k_hist2(const float* s, const int* meta, int* hist, long npix){
  long i = (long)blockIdx.x * TPB + threadIdx.x;
  if (i >= 2 * npix) return;
  int b = (int)(i / npix);
  unsigned u = __float_as_uint(s[i]);
  if ((int)(u >> 24) == meta[b * 8 + 0])
    atomicAdd(&hist[b * 256 + (int)((u >> 16) & 255)], 1);
}
__global__ void k_thresh2(const int* hist, int* meta, int Ksel){
  int b = blockIdx.x;
  const int* h = hist + b * 256;
  int R = Ksel - meta[b * 8 + 1];
  int cum = 0, T = 0;
  for (int bin = 255; bin >= 0; bin--){
    int c = h[bin];
    if (cum + c >= R){ T = bin; break; }
    cum += c;
  }
  meta[b * 8 + 2] = T;
  int selHi = meta[b * 8 + 1] + cum;
  meta[b * 8 + 3] = selHi;
  meta[b * 8 + 4] = Ksel - selHi;
}
__global__ void k_collect(const float* s, const int* meta, int* cnt,
                          float* vals, int* idxs, long npix, int Ksel){
  long i = (long)blockIdx.x * TPB + threadIdx.x;
  if (i >= 2 * npix) return;
  int b = (int)(i / npix); int p = (int)(i % npix);
  float v = s[i];
  unsigned u = __float_as_uint(v);
  int T1 = meta[b * 8 + 0], T2 = meta[b * 8 + 2];
  int selHi = meta[b * 8 + 3], R2 = meta[b * 8 + 4];
  int k1 = (int)(u >> 24), k2 = (int)((u >> 16) & 255);
  int pos = -1;
  if (k1 > T1 || (k1 == T1 && k2 > T2)) pos = atomicAdd(&cnt[b * 2 + 0], 1);
  else if (k1 == T1 && k2 == T2){
    int q = atomicAdd(&cnt[b * 2 + 1], 1);
    if (q < R2) pos = selHi + q;
  }
  if (pos >= 0 && pos < Ksel){ vals[b * Ksel + pos] = v; idxs[b * Ksel + pos] = p; }
}
__global__ __launch_bounds__(1024) void k_topk_sort(float* vals, int* idxs){
  int b = blockIdx.x, t = threadIdx.x;
  __shared__ float v[1024]; __shared__ int id[1024];
  v[t] = vals[b * 1024 + t]; id[t] = idxs[b * 1024 + t];
  __syncthreads();
  for (int size = 2; size <= 1024; size <<= 1){
    for (int stride = size >> 1; stride > 0; stride >>= 1){
      int j = t ^ stride;
      if (j > t){
        bool desc = ((t & size) == 0);
        float va = v[t], vb = v[j]; int ia = id[t], ib = id[j];
        bool a_first = (va > vb) || (va == vb && ia < ib);
        bool sw = desc ? !a_first : a_first;
        if (sw){ v[t] = vb; v[j] = va; id[t] = ib; id[j] = ia; }
      }
      __syncthreads();
    }
  }
  vals[b * 1024 + t] = v[t]; idxs[b * 1024 + t] = id[t];
}

__global__ void k_vfeat(const float* vals, const int* idxs, _Float16* vf, int Wf, int Hf){
  int i = blockIdx.x * TPB + threadIdx.x;
  if (i >= 2048) return;
  int idv = idxs[i];
  int row = idv / Wf, col = idv % Wf;
  float xn = ((float)col - Wf * 0.5f + 0.5f) / (float)Wf;
  float yn = ((float)row - Hf * 0.5f + 0.5f) / (float)Hf;
  _Float16* o = vf + (long)i * 32;
  o[0] = (_Float16)xn; o[1] = (_Float16)yn; o[2] = (_Float16)vals[i];
  for (int c = 3; c < 32; c++) o[c] = (_Float16)0.f;
}
__global__ void k_dtgather(const float* dist, const int* idxs, _Float16* dt){
  long t = (long)blockIdx.x * TPB + threadIdx.x;
  if (t >= (long)2048 * 192) return;
  int i = (int)(t / 192), f = (int)(t % 192);
  int b = i >> 10;
  int idv = idxs[i];
  int row = idv / 800, col = idv % 800;
  int c = f / 64, r = f % 64;
  int di = r / 8, dj = r % 8;
  int gr = (row / 8) * 8 + di, gc = (col / 8) * 8 + dj;
  float v = dist[(((long)(b * 3 + c)) * 400 + gr) * 800 + gc] * 0.1f;
  dt[(long)i * 192 + f] = (_Float16)v;
}

// ======================= BN / elementwise =======================
__global__ __launch_bounds__(TPB) void k_bn_stats(const float* X, float* mv, int N, int M){
  int c = blockIdx.x;
  __shared__ float ss[TPB], sq[TPB];
  float a = 0.f, q = 0.f;
  for (int n = threadIdx.x; n < N; n += TPB){ float v = X[(long)n * M + c]; a += v; q += v * v; }
  ss[threadIdx.x] = a; sq[threadIdx.x] = q; __syncthreads();
  for (int s2 = TPB / 2; s2 > 0; s2 >>= 1){
    if (threadIdx.x < s2){ ss[threadIdx.x] += ss[threadIdx.x + s2]; sq[threadIdx.x] += sq[threadIdx.x + s2]; }
    __syncthreads();
  }
  if (threadIdx.x == 0){
    float m = ss[0] / N;
    mv[c] = m; mv[M + c] = sq[0] / N - m * m;
  }
}
__global__ void k_bn_relu(const float* X, const float* mv, const float* g, const float* be,
                          _Float16* Y, long total, int M){
  long i = (long)blockIdx.x * TPB + threadIdx.x;
  if (i >= total) return;
  int c = (int)(i % M);
  float v = (X[i] - mv[c]) * rsqrtf(mv[M + c] + 1e-5f) * g[c] + be[c];
  Y[i] = (_Float16)fmaxf(v, 0.f);
}
__global__ void k_add_cvt(const float* A, const float* B, float* O32, _Float16* O16, long n){
  long i = (long)blockIdx.x * TPB + threadIdx.x;
  if (i >= n) return;
  float v = A[i] + B[i];
  O32[i] = v; O16[i] = (_Float16)v;
}
__global__ void k_cvt_h(const float* src, _Float16* dst, long n){
  long i = (long)blockIdx.x * TPB + threadIdx.x;
  if (i >= n) return;
  dst[i] = (_Float16)src[i];
}

// ======================= attention repacks =======================
__global__ void k_repack_nd(const float* __restrict__ src, _Float16* __restrict__ dst){
  int t = blockIdx.x * TPB + threadIdx.x;
  if (t >= 524288) return;                          // [b][h][n][d], d<64, n<1024, h<4, b<2
  int d = t & 63, n = (t >> 6) & 1023, h = (t >> 16) & 3, b = t >> 18;
  dst[t] = (_Float16)src[((long)(b * 1024 + n)) * 256 + d * 4 + h];
}
__global__ void k_repack_v(const float* __restrict__ src, _Float16* __restrict__ dst){
  int t = blockIdx.x * TPB + threadIdx.x;
  if (t >= 524288) return;                          // [b][h][d][m]
  int m = t & 1023, d = (t >> 10) & 63, h = (t >> 16) & 3, b = t >> 18;
  dst[t] = (_Float16)src[((long)(b * 1024 + m)) * 256 + d * 4 + h];
}
__global__ void k_repack_msg(const float* __restrict__ src, _Float16* __restrict__ dst){
  int t = blockIdx.x * TPB + threadIdx.x;
  if (t >= 524288) return;
  int d = t & 63, n = (t >> 6) & 1023, h = (t >> 16) & 3, b = t >> 18;
  dst[((long)(b * 1024 + n)) * 256 + d * 4 + h] = (_Float16)src[t];
}
__global__ __launch_bounds__(TPB) void k_row_softmax(const float* S, _Float16* P, int cols, float sc){
  long row = blockIdx.x;
  const float* r = S + row * (long)cols;
  _Float16*    p = P + row * (long)cols;
  __shared__ float red[TPB];
  float mx = -3.4e38f;
  for (int j = threadIdx.x; j < cols; j += TPB) mx = fmaxf(mx, r[j] * sc);
  red[threadIdx.x] = mx; __syncthreads();
  for (int s2 = TPB / 2; s2 > 0; s2 >>= 1){
    if (threadIdx.x < s2) red[threadIdx.x] = fmaxf(red[threadIdx.x], red[threadIdx.x + s2]);
    __syncthreads();
  }
  mx = red[0]; __syncthreads();
  float sm = 0.f;
  for (int j = threadIdx.x; j < cols; j += TPB) sm += expf(r[j] * sc - mx);
  red[threadIdx.x] = sm; __syncthreads();
  for (int s2 = TPB / 2; s2 > 0; s2 >>= 1){
    if (threadIdx.x < s2) red[threadIdx.x] += red[threadIdx.x + s2];
    __syncthreads();
  }
  float inv = 1.f / red[0];
  for (int j = threadIdx.x; j < cols; j += TPB) p[j] = (_Float16)(expf(r[j] * sc - mx) * inv);
}
__global__ void k_cat(const _Float16* X16, const float* Mrg, _Float16* Cat, long total){
  long i = (long)blockIdx.x * TPB + threadIdx.x;
  if (i >= total) return;
  int c = (int)(i & 511); long n = i >> 9;
  Cat[i] = (c < 256) ? X16[n * 256 + c] : (_Float16)Mrg[n * 256 + (c - 256)];
}

// ======================= Sinkhorn =======================
__global__ void k_buildC(const float* SC, const float* alpha, float* Cm, long total){
  long i = (long)blockIdx.x * TPB + threadIdx.x;
  if (i >= total) return;
  long per = 1025L * 1025L;
  int b = (int)(i / per); long r = i % per;
  int ii = (int)(r / 1025), jj = (int)(r % 1025);
  Cm[i] = (ii < 1024 && jj < 1024) ? SC[((long)b * 1024 + ii) * 1024 + jj] : alpha[0];
}
__global__ __launch_bounds__(TPB) void k_row_lse(const float* C, const float* v, float* u){
  int b = blockIdx.x / 1025, ii = blockIdx.x % 1025;
  const float* row = C + (long)blockIdx.x * 1025;
  const float* vb  = v + b * 1025;
  __shared__ float red[TPB];
  float mx = -3.4e38f;
  for (int j = threadIdx.x; j < 1025; j += TPB) mx = fmaxf(mx, row[j] + vb[j]);
  red[threadIdx.x] = mx; __syncthreads();
  for (int s2 = TPB / 2; s2 > 0; s2 >>= 1){
    if (threadIdx.x < s2) red[threadIdx.x] = fmaxf(red[threadIdx.x], red[threadIdx.x + s2]);
    __syncthreads();
  }
  mx = red[0]; __syncthreads();
  float sm = 0.f;
  for (int j = threadIdx.x; j < 1025; j += TPB) sm += expf(row[j] + vb[j] - mx);
  red[threadIdx.x] = sm; __syncthreads();
  for (int s2 = TPB / 2; s2 > 0; s2 >>= 1){
    if (threadIdx.x < s2) red[threadIdx.x] += red[threadIdx.x + s2];
    __syncthreads();
  }
  if (threadIdx.x == 0){
    const float norm = -logf(2048.f);
    float lmu = (ii < 1024) ? norm : (logf(1024.f) + norm);
    u[b * 1025 + ii] = lmu - (logf(red[0]) + mx);
  }
}
__global__ __launch_bounds__(TPB) void k_col_lse(const float* C, const float* u, float* v){
  int b = blockIdx.x / 1025, jj = blockIdx.x % 1025;
  const float* ub   = u + b * 1025;
  const float* base = C + (long)b * 1025 * 1025 + jj;
  __shared__ float red[TPB];
  float mx = -3.4e38f;
  for (int i = threadIdx.x; i < 1025; i += TPB) mx = fmaxf(mx, base[(long)i * 1025] + ub[i]);
  red[threadIdx.x] = mx; __syncthreads();
  for (int s2 = TPB / 2; s2 > 0; s2 >>= 1){
    if (threadIdx.x < s2) red[threadIdx.x] = fmaxf(red[threadIdx.x], red[threadIdx.x + s2]);
    __syncthreads();
  }
  mx = red[0]; __syncthreads();
  float sm = 0.f;
  for (int i = threadIdx.x; i < 1025; i += TPB) sm += expf(base[(long)i * 1025] + ub[i] - mx);
  red[threadIdx.x] = sm; __syncthreads();
  for (int s2 = TPB / 2; s2 > 0; s2 >>= 1){
    if (threadIdx.x < s2) red[threadIdx.x] += red[threadIdx.x + s2];
    __syncthreads();
  }
  if (threadIdx.x == 0){
    const float norm = -logf(2048.f);
    float lnu = (jj < 1024) ? norm : (logf(1024.f) + norm);
    v[b * 1025 + jj] = lnu - (logf(red[0]) + mx);
  }
}
__global__ void k_writeZ(const float* C, const float* u, const float* v, float* Z, long total){
  long i = (long)blockIdx.x * TPB + threadIdx.x;
  if (i >= total) return;
  long per = 1025L * 1025L;
  int b = (int)(i / per); long r = i % per;
  int ii = (int)(r / 1025), jj = (int)(r % 1025);
  Z[i] = C[i] + u[b * 1025 + ii] + v[b * 1025 + jj] + logf(2048.f);
}
__global__ void k_cls(const float* X32, const float* Wc, const float* bc, float* out){
  int i = blockIdx.x * TPB + threadIdx.x;
  if (i >= 6144) return;
  int k = i & 1023; int bcid = i >> 10;
  int c = bcid % 3, b = bcid / 3;
  const float* x = X32 + ((long)(b * 1024 + k)) * 256;
  float acc = bc[c];
  for (int o = 0; o < 256; o++) acc += Wc[c * 256 + o] * x[o];
  out[((long)(b * 3 + c)) * 1024 + k] = acc;
}

// ======================= host orchestration =======================
struct Lyr { const _Float16* w; const float* b; const float* g; const float* be; int M; int Kp; };

extern "C" void kernel_launch(void* const* d_in, const int* in_sizes, int n_in,
                              void* d_out, int out_size, void* d_ws, size_t ws_size,
                              hipStream_t stream)
{
  (void)in_sizes; (void)n_in; (void)out_size; (void)ws_size;
  const int hc = 50, wc = 100, Hf = 400, Wf = 800;
  const long NPIX = (long)Hf * Wf;     // 320000
  const int  Kv = 1024, NPOS = 2048;

  char* WS = (char*)d_ws;
  size_t off = 0;
  auto alloc = [&](size_t bytes)->char*{
    char* p = WS + off; off = (off + bytes + 255) & ~(size_t)255; return p;
  };

  float* s    = (float*)alloc(2 * NPIX * 4);
  float* t1   = (float*)alloc(2 * NPIX * 4);
  float* t2   = (float*)alloc(2 * NPIX * 4);
  float* t3   = (float*)alloc(2 * NPIX * 4);
  float* mask = (float*)alloc(2 * NPIX * 4);
  float* ssb  = (float*)alloc(2 * NPIX * 4);
  char*  hblk = alloc(8192);
  int* hist1 = (int*)hblk;
  int* hist2 = (int*)(hblk + 2048);
  int* cnt   = (int*)(hblk + 4096);
  int* meta  = (int*)(hblk + 4160);
  float* vals = (float*)alloc(2 * Kv * 4);
  int*   idxs = (int*)  alloc(2 * Kv * 4);
  _Float16* vf16 = (_Float16*)alloc((size_t)NPOS * 32 * 2);
  _Float16* dt16 = (_Float16*)alloc((size_t)NPOS * 192 * 2);
  _Float16* E16a = (_Float16*)alloc((size_t)NPOS * 512 * 2);
  _Float16* E16b = (_Float16*)alloc((size_t)NPOS * 512 * 2);
  float*    G32  = (float*)   alloc((size_t)NPOS * 512 * 4);
  float*    X32  = (float*)   alloc((size_t)NPOS * 256 * 4);
  _Float16* X16  = (_Float16*)alloc((size_t)NPOS * 256 * 2);
  float*    mv   = (float*)   alloc(1024 * 4);
  _Float16* Qh   = (_Float16*)alloc((size_t)8 * 1024 * 64 * 2);
  _Float16* Kh   = (_Float16*)alloc((size_t)8 * 1024 * 64 * 2);
  _Float16* Vh   = (_Float16*)alloc((size_t)8 * 1024 * 64 * 2);
  float*    S32  = (float*)   alloc((size_t)8 * 1024 * 1024 * 4);
  _Float16* P16  = (_Float16*)alloc((size_t)8 * 1024 * 1024 * 2);
  float*    Mh32 = (float*)   alloc((size_t)8 * 1024 * 64 * 4);
  _Float16* Msg16= (_Float16*)alloc((size_t)NPOS * 256 * 2);
  _Float16* Cat16= (_Float16*)alloc((size_t)NPOS * 512 * 2);
  _Float16* D16  = (_Float16*)alloc((size_t)NPOS * 256 * 2);
  float*    uv   = (float*)   alloc(2 * 2 * 1025 * 4);
  float* u_ = uv; float* v_ = uv + 2 * 1025;
  float* SC32 = S32;            // reuse (8.4MB of 32MB)
  float* Cmat = (float*)P16;    // reuse (8.4MB of 16MB)

  auto F = [&](int i)->const float*{ return (const float*)d_in[i]; };
  auto mkW16 = [&](int leaf, int M, int K)->const _Float16*{
    int Kp = (K + 31) & ~31;
    _Float16* dst = (_Float16*)alloc((size_t)M * Kp * 2);
    k_cvt_w16<<<G((long)M * Kp), TPB, 0, stream>>>(F(leaf), dst, M, K, Kp);
    return dst;
  };
  auto gemm = [&](const _Float16* A, const _Float16* B, float* C, const float* bias,
                  int M, int Kd, int N, float cscale,
                  long long sA, long long sB, long long sC, int batch){
    long units = ((long)(M >> 4)) * (N >> 6);
    dim3 grid((unsigned)((units + 7) / 8), (unsigned)batch, 1);
    k_wmma_gemm<<<grid, 256, 0, stream>>>(A, B, C, bias, M, Kd, N, cscale, sA, sB, sC);
  };

  // ---- weight tables (leaf order per setup_inputs pytree flattening) ----
  Lyr venc[5]; { int vc[6] = {3,32,64,128,256,256}; int li = 2;
    for (int l = 0; l < 5; l++){
      int M = vc[l+1], K = vc[l];
      venc[l].w = mkW16(li, M, K); venc[l].b = F(li + 1);
      if (l < 4){ venc[l].g = F(li + 2); venc[l].be = F(li + 3); li += 4; }
      else      { venc[l].g = nullptr;  venc[l].be = nullptr;   li += 2; }
      venc[l].M = M; venc[l].Kp = (K + 31) & ~31;
    } }
  Lyr dtenc[4]; { int dc[5] = {192,64,128,256,256}; int li = 20;
    for (int l = 0; l < 4; l++){
      int M = dc[l+1], K = dc[l];
      dtenc[l].w = mkW16(li, M, K); dtenc[l].b = F(li + 1);
      if (l < 3){ dtenc[l].g = F(li + 2); dtenc[l].be = F(li + 3); li += 4; }
      else      { dtenc[l].g = nullptr;  dtenc[l].be = nullptr;   li += 2; }
      dtenc[l].M = M; dtenc[l].Kp = (K + 31) & ~31;
    } }
  struct GnnW { const _Float16 *wq,*wk,*wv,*wm,*w1,*w2;
                const float *bq,*bk,*bv,*bm,*b1,*g1,*be1,*b2; };
  GnnW gw[7];
  for (int l = 0; l < 7; l++){
    int t = 34 + 14 * l;
    gw[l].wq = mkW16(t + 0, 256, 256); gw[l].bq = F(t + 1);
    gw[l].wk = mkW16(t + 2, 256, 256); gw[l].bk = F(t + 3);
    gw[l].wv = mkW16(t + 4, 256, 256); gw[l].bv = F(t + 5);
    gw[l].wm = mkW16(t + 6, 256, 256); gw[l].bm = F(t + 7);
    gw[l].w1 = mkW16(t + 8, 512, 512); gw[l].b1 = F(t + 9);
    gw[l].g1 = F(t + 10); gw[l].be1 = F(t + 11);
    gw[l].w2 = mkW16(t + 12, 256, 512); gw[l].b2 = F(t + 13);
  }
  const _Float16* wfinal = mkW16(132, 256, 256);
  const float* bfinal = F(133);
  const float* alpha  = F(134);
  const float* clsW   = F(135);
  const float* clsb   = F(136);

  // ---- frontend: cell softmax + unshuffle + NMS ----
  const float* vertex = F(0);
  const float* dist   = F(1);
  long n2 = 2 * NPIX;
  k_cell_softmax<<<G(2L * hc * wc), TPB, 0, stream>>>(vertex, s, hc, wc);
  k_mp_row<<<G(n2), TPB, 0, stream>>>(s, t1, Hf, Wf);
  k_mp_col<<<G(n2), TPB, 0, stream>>>(t1, t2, Hf, Wf);
  k_eq_mask<<<G(n2), TPB, 0, stream>>>(s, t2, mask, n2);
  for (int it = 0; it < 2; it++){
    k_mp_row<<<G(n2), TPB, 0, stream>>>(mask, t1, Hf, Wf);
    k_mp_col<<<G(n2), TPB, 0, stream>>>(t1, t2, Hf, Wf);            // t2 = pooled mask
    k_suppress<<<G(n2), TPB, 0, stream>>>(t2, s, ssb, n2);
    k_mp_row<<<G(n2), TPB, 0, stream>>>(ssb, t1, Hf, Wf);
    k_mp_col<<<G(n2), TPB, 0, stream>>>(t1, t3, Hf, Wf);            // t3 = pooled ss
    k_mask_update<<<G(n2), TPB, 0, stream>>>(mask, ssb, t3, t2, n2);
  }
  k_apply_mask<<<G(n2), TPB, 0, stream>>>(s, mask, n2);

  // ---- top-K select + sort ----
  (void)hipMemsetAsync(hblk, 0, 8192, stream);
  k_hist1<<<G(n2), TPB, 0, stream>>>(s, hist1, NPIX);
  k_thresh1<<<2, 1, 0, stream>>>(hist1, meta, Kv);
  k_hist2<<<G(n2), TPB, 0, stream>>>(s, meta, hist2, NPIX);
  k_thresh2<<<2, 1, 0, stream>>>(hist2, meta, Kv);
  k_collect<<<G(n2), TPB, 0, stream>>>(s, meta, cnt, vals, idxs, NPIX, Kv);
  k_topk_sort<<<2, 1024, 0, stream>>>(vals, idxs);

  // ---- feature gathers ----
  k_vfeat<<<G(2048), TPB, 0, stream>>>(vals, idxs, vf16, Wf, Hf);
  k_dtgather<<<G((long)NPOS * 192), TPB, 0, stream>>>(dist, idxs, dt16);

  // ---- encoder MLP chains (WMMA GEMM + BN(batch-stats) + ReLU) ----
  auto run_chain = [&](const Lyr* L, int nL, const _Float16* in0, float* out32){
    const _Float16* cur = in0;
    _Float16* ping[2] = {E16a, E16b}; int tg = 0;
    for (int l = 0; l < nL; l++){
      float* o = (l == nL - 1) ? out32 : G32;
      gemm(L[l].w, cur, o, L[l].b, L[l].M, L[l].Kp, NPOS, 1.f, 0, 0, 0, 1);
      if (l < nL - 1){
        long tot = (long)NPOS * L[l].M;
        k_bn_stats<<<L[l].M, TPB, 0, stream>>>(o, mv, NPOS, L[l].M);
        k_bn_relu<<<G(tot), TPB, 0, stream>>>(o, mv, L[l].g, L[l].be, ping[tg], tot, L[l].M);
        cur = ping[tg]; tg ^= 1;
      }
    }
  };
  run_chain(venc, 5, vf16, X32);
  run_chain(dtenc, 4, dt16, G32);
  k_add_cvt<<<G((long)NPOS * 256), TPB, 0, stream>>>(X32, G32, X32, X16, (long)NPOS * 256);

  // ---- 7 attentional GNN layers ----
  for (int l = 0; l < 7; l++){
    gemm(gw[l].wq, X16, G32, gw[l].bq, 256, 256, NPOS, 1.f, 0, 0, 0, 1);
    k_repack_nd<<<G(524288), TPB, 0, stream>>>(G32, Qh);
    gemm(gw[l].wk, X16, G32, gw[l].bk, 256, 256, NPOS, 1.f, 0, 0, 0, 1);
    k_repack_nd<<<G(524288), TPB, 0, stream>>>(G32, Kh);
    gemm(gw[l].wv, X16, G32, gw[l].bv, 256, 256, NPOS, 1.f, 0, 0, 0, 1);
    k_repack_v<<<G(524288), TPB, 0, stream>>>(G32, Vh);
    // S[n][m] = sum_d Qh[n][d]*Kh[m][d], batched over 8 (b,h)
    gemm(Kh, Qh, S32, nullptr, 1024, 64, 1024, 1.f, 65536LL, 65536LL, 1048576LL, 8);
    k_row_softmax<<<8192, TPB, 0, stream>>>(S32, P16, 1024, 0.125f);
    // Mh[n][d] = sum_m P[n][m]*Vh[d][m]
    gemm(Vh, P16, Mh32, nullptr, 64, 1024, 1024, 1.f, 65536LL, 1048576LL, 65536LL, 8);
    k_repack_msg<<<G(524288), TPB, 0, stream>>>(Mh32, Msg16);
    gemm(gw[l].wm, Msg16, G32, gw[l].bm, 256, 256, NPOS, 1.f, 0, 0, 0, 1);
    k_cat<<<G((long)NPOS * 512), TPB, 0, stream>>>(X16, G32, Cat16, (long)NPOS * 512);
    gemm(gw[l].w1, Cat16, G32, gw[l].b1, 512, 512, NPOS, 1.f, 0, 0, 0, 1);
    k_bn_stats<<<512, TPB, 0, stream>>>(G32, mv, NPOS, 512);
    k_bn_relu<<<G((long)NPOS * 512), TPB, 0, stream>>>(G32, mv, gw[l].g1, gw[l].be1,
                                                       E16a, (long)NPOS * 512, 512);
    gemm(gw[l].w2, E16a, G32, gw[l].b2, 256, 512, NPOS, 1.f, 0, 0, 0, 1);
    k_add_cvt<<<G((long)NPOS * 256), TPB, 0, stream>>>(X32, G32, X32, X16, (long)NPOS * 256);
  }

  // ---- classification head (on post-GNN embedding) ----
  float* Zout = (float*)d_out;
  k_cls<<<G(6144), TPB, 0, stream>>>(X32, clsW, clsb, Zout + 2 * 1025L * 1025L);

  // ---- final descriptors + score Gram matrix + Sinkhorn ----
  gemm(wfinal, X16, G32, bfinal, 256, 256, NPOS, 1.f, 0, 0, 0, 1);
  k_cvt_h<<<G((long)NPOS * 256), TPB, 0, stream>>>(G32, D16, (long)NPOS * 256);
  gemm(D16, D16, SC32, nullptr, 1024, 256, 1024, 0.0625f, 262144LL, 262144LL, 1048576LL, 2);
  long ctot = 2 * 1025L * 1025L;
  k_buildC<<<G(ctot), TPB, 0, stream>>>(SC32, alpha, Cmat, ctot);
  (void)hipMemsetAsync(uv, 0, 2 * 2 * 1025 * 4, stream);
  for (int it = 0; it < 20; it++){
    k_row_lse<<<2 * 1025, TPB, 0, stream>>>(Cmat, v_, u_);
    k_col_lse<<<2 * 1025, TPB, 0, stream>>>(Cmat, u_, v_);
  }
  k_writeZ<<<G(ctot), TPB, 0, stream>>>(Cmat, u_, v_, Zout, ctot);
}
